// CausalSelfAttentionARMA_25598005084331
// MI455X (gfx1250) — matrix operand order repaired and checked by reading, not compiled, not run on hardware
//
#include <hip/hip_runtime.h>

// ---------------------------------------------------------------------------
// CDNA5 (gfx1250) wave32 WMMA implementation of CausalSelfAttentionARMA.
// Matmuls: v_wmma_f32_16x16x32_bf16. GEMM B tiles are staged to LDS with
// global_load_async_to_lds_b128 (double-buffered, s_wait_asynccnt).
// ---------------------------------------------------------------------------

typedef __attribute__((ext_vector_type(16))) __bf16 v16bf;
typedef __attribute__((ext_vector_type(8)))  __bf16 v8bf;
typedef __attribute__((ext_vector_type(8)))  float  v8f;

#define DEV __device__ __forceinline__

static constexpr int Bc = 2;
static constexpr int Tc = 2048;
static constexpr int Cc = 1024;
static constexpr int Hc = 16;
static constexpr int Dc = 64;

DEV v8f wmma_bf16(v16bf a, v16bf b, v8f c) {
  // (neg_a, A, neg_b, B, c_mod, C, reuse_a, reuse_b)
  return __builtin_amdgcn_wmma_f32_16x16x32_bf16(false, a, false, b, (short)0, c,
                                                 false, false);
}

DEV v16bf cat8(v8bf lo, v8bf hi) {
  v16bf r;
#pragma unroll
  for (int i = 0; i < 8; ++i) { r[i] = lo[i]; r[i + 8] = hi[i]; }
  return r;
}

// A-fragment (16x32 bf16): lane holds row = lane%16; K chunk at
// hf*16 .. hf*16+15  (hf = lane/16).
DEV v16bf load_afrag(const float* row, int hf) {
  const float* p0 = row + hf * 16;
  v8f a0 = *(const v8f*)p0;          // 32B aligned
  v8f a1 = *(const v8f*)(p0 + 8);
  v16bf r;
#pragma unroll
  for (int i = 0; i < 8; ++i) { r[i] = (__bf16)a0[i]; r[i + 8] = (__bf16)a1[i]; }
  return r;
}

DEV v16bf load_afrag(const __bf16* row, int hf) {
  const __bf16* p0 = row + hf * 16;
  return cat8(*(const v8bf*)p0, *(const v8bf*)(p0 + 8));
}

// Async-copy one 64x32 bf16 B tile (4KB) from WT[rowbase..rowbase+63][k..k+31]
// into LDS at ldsbase. 128 threads x 32B each, 2 async b128 per thread.
// LDS layout: row r at byte r*64, k-halves of 16 elems at +0 / +32.
DEV void stage_b_tile(const __bf16* WT, size_t rowbase, int K, int k,
                      unsigned ldsbase, int tid) {
  const __bf16* src = WT + (rowbase + (size_t)(tid >> 1)) * (size_t)K + k + (tid & 1) * 16;
  unsigned long long g = (unsigned long long)(size_t)src;
  unsigned l = ldsbase + (unsigned)tid * 32u;
  asm volatile("global_load_async_to_lds_b128 %0, %1, off"
               :: "v"(l), "v"(g) : "memory");
  asm volatile("global_load_async_to_lds_b128 %0, %1, off offset:16"
               :: "v"(l), "v"(g) : "memory");
}

// ---------------------------------------------------------------------------
// GEMM: 4 waves per workgroup, 64x64 tile (each wave 16 rows x 64 cols).
// B tile shared through LDS, double-buffered with async copies.
// MODE 0: split/scatter into q_bf (scaled 1/8) and k_bf  (B,H,T,d) bf16
// MODE 1: scatter into k2_bf (B,H,T,d) bf16, row t==T-1 forced to 0
// MODE 2: fp32 out[row*N+col] = acc + 2*bias
// ---------------------------------------------------------------------------
template <typename AT, int MODE>
__global__ void gemm_wmma_kernel(const AT* __restrict__ A,
                                 const __bf16* __restrict__ WT,
                                 const float* __restrict__ bias,
                                 void* __restrict__ out0, void* __restrict__ out1,
                                 int K, int N) {
  __shared__ __bf16 Bt[2][64 * 32];       // 2 x 4KB
  const int tid = threadIdx.x;            // 0..127
  const int wv = tid >> 5;
  const int lane = tid & 31;
  const int hf = lane >> 4;
  const int lm = lane & 15;
  const int mb = blockIdx.x, nb = blockIdx.y;

  const v8f vz = {0.f, 0.f, 0.f, 0.f, 0.f, 0.f, 0.f, 0.f};
  v8f acc[4] = {vz, vz, vz, vz};

  const AT* arow = A + (size_t)(mb * 64 + wv * 16 + lm) * K;
  const size_t rowbase = (size_t)nb * 64;
  const unsigned lds0 = (unsigned)(size_t)&Bt[0][0];
  const unsigned lds1 = (unsigned)(size_t)&Bt[1][0];

  const int nk = K / 32;
  stage_b_tile(WT, rowbase, K, 0, lds0, tid);
  for (int kc = 0; kc < nk; ++kc) {
    const int k = kc * 32;
    if (kc + 1 < nk) {
      stage_b_tile(WT, rowbase, K, k + 32, ((kc + 1) & 1) ? lds1 : lds0, tid);
      asm volatile("s_wait_asynccnt 0x2" ::: "memory");   // current buffer done
    } else {
      asm volatile("s_wait_asynccnt 0x0" ::: "memory");
    }
    __syncthreads();                       // all waves' copies landed

    if (k + 32 < K) __builtin_prefetch(arow + k + 32, 0, 1);
    v16bf af = load_afrag(arow + k, hf);
    const __bf16* bbase = &Bt[kc & 1][0];
#pragma unroll
    for (int nt = 0; nt < 4; ++nt) {
      const __bf16* brow = bbase + (nt * 16 + lm) * 32 + hf * 16;
      acc[nt] = wmma_bf16(af, *(const v16bf*)brow, acc[nt]);
    }
    __syncthreads();                       // reads done before buffer reused
  }

#pragma unroll
  for (int nt = 0; nt < 4; ++nt) {
    const int col = nb * 64 + nt * 16 + lm;
    const float bv = bias[col];
#pragma unroll
    for (int r = 0; r < 8; ++r) {
      const int row = mb * 64 + wv * 16 + r + 8 * hf;   // row = b*T + t
      const int b = row / Tc, t = row % Tc;
      const float val = acc[nt][r] + bv;
      if (MODE == 0) {
        if (col < Cc) {
          const int h = col / Dc, dd = col % Dc;
          ((__bf16*)out0)[(((size_t)(b * Hc + h)) * Tc + t) * Dc + dd] =
              (__bf16)(val * 0.125f);                    // q pre-scaled 1/sqrt(d)
        } else {
          const int c2 = col - Cc;
          const int h = c2 / Dc, dd = c2 % Dc;
          ((__bf16*)out1)[(((size_t)(b * Hc + h)) * Tc + t) * Dc + dd] = (__bf16)val;
        }
      } else if (MODE == 1) {
        const int h = col / Dc, dd = col % Dc;
        ((__bf16*)out0)[(((size_t)(b * Hc + h)) * Tc + t) * Dc + dd] =
            (__bf16)((t < Tc - 1) ? val : 0.f);          // zero the pad row
      } else {
        ((float*)out0)[(size_t)row * N + col] = acc[nt][r] + 2.f * bv;
      }
    }
  }
}

// ---------------------------------------------------------------------------
// Flash-attention: one wave per (head, 16-row q block). Q,K row-major (T,d)
// bf16 per head; V supplied transposed (d,T) bf16 per head. Online softmax
// in fp32; P tile goes through LDS to re-layout C-frag -> A-frag for P*V.
// ---------------------------------------------------------------------------
__global__ void flash_attn_kernel(const __bf16* __restrict__ Q,
                                  const __bf16* __restrict__ Kmat,
                                  const __bf16* __restrict__ Vt,
                                  float* __restrict__ Y,
                                  int Tq, int Tk) {
  __shared__ __bf16 Plds[16 * 48];          // 16x32 P tile, padded stride 48
  const int lane = threadIdx.x;
  const int hf = lane >> 4;
  const int lm = lane & 15;
  const int head = blockIdx.y;
  const int qrow0 = blockIdx.x * 16;

  const __bf16* Qh = Q + (size_t)head * Tc * Dc;
  const __bf16* Kh = Kmat + (size_t)head * Tc * Dc;
  const __bf16* Vh = Vt + (size_t)head * Dc * Tc;
  float* Yh = Y + (size_t)head * Tc * Dc;

  const __bf16* qrow = Qh + (size_t)(qrow0 + lm) * Dc;
  const v16bf qf0 = cat8(*(const v8bf*)(qrow + hf * 16),
                         *(const v8bf*)(qrow + hf * 16 + 8));
  const v16bf qf1 = cat8(*(const v8bf*)(qrow + 32 + hf * 16),
                         *(const v8bf*)(qrow + 32 + hf * 16 + 8));

  const v8f vz = {0.f, 0.f, 0.f, 0.f, 0.f, 0.f, 0.f, 0.f};
  v8f acc[4] = {vz, vz, vz, vz};
  float mrow[8], lrow[8];
#pragma unroll
  for (int r = 0; r < 8; ++r) { mrow[r] = -1e30f; lrow[r] = 0.f; }

  const int nchunks = (qrow0 >> 5) + 1;     // causal: cover cols <= qrow0+15
  for (int c = 0; c < nchunks; ++c) {
    const int col0 = c * 32;
    if (c + 1 < nchunks)
      __builtin_prefetch(Kh + (size_t)(col0 + 32 + lm) * Dc, 0, 1);

    // S = Q K^T for 16q x 32k  (2 N-tiles x 2 K-chunks of 32 over d=64)
    v8f s0 = vz, s1 = vz;
    {
      const __bf16* k0 = Kh + (size_t)(col0 + lm) * Dc + hf * 16;
      const __bf16* k1 = Kh + (size_t)(col0 + 16 + lm) * Dc + hf * 16;
      s0 = wmma_bf16(qf0, *(const v16bf*)k0, s0);
      s0 = wmma_bf16(qf1, *(const v16bf*)(k0 + 32), s0);
      s1 = wmma_bf16(qf0, *(const v16bf*)k1, s1);
      s1 = wmma_bf16(qf1, *(const v16bf*)(k1 + 32), s1);
    }

    // online softmax (row = qrow0 + r + 8*hf ; col = col0 + lm / +16)
    float p0v[8], p1v[8];
#pragma unroll
    for (int r = 0; r < 8; ++r) {
      const int grow = qrow0 + r + 8 * hf;
      const int c0 = col0 + lm, c1 = col0 + 16 + lm;
      float sa = ((c0 <= grow) && (c0 < Tk)) ? s0[r] : -1e30f;
      float sb = ((c1 <= grow) && (c1 < Tk)) ? s1[r] : -1e30f;
      float mx = fmaxf(sa, sb);
#pragma unroll
      for (int msk = 1; msk < 16; msk <<= 1)
        mx = fmaxf(mx, __shfl_xor(mx, msk, 32));
      const float mnew = fmaxf(mrow[r], mx);
      const float alpha = __expf(mrow[r] - mnew);
      const float e0 = __expf(sa - mnew);
      const float e1 = __expf(sb - mnew);
      float rs = e0 + e1;
#pragma unroll
      for (int msk = 1; msk < 16; msk <<= 1) rs += __shfl_xor(rs, msk, 32);
      lrow[r] = lrow[r] * alpha + rs;
      mrow[r] = mnew;
      p0v[r] = e0; p1v[r] = e1;
#pragma unroll
      for (int nt = 0; nt < 4; ++nt) acc[nt][r] *= alpha;
    }

    // C-layout -> A-layout via LDS
    __syncthreads();
#pragma unroll
    for (int r = 0; r < 8; ++r) {
      const int prow = r + 8 * hf;
      Plds[prow * 48 + lm] = (__bf16)p0v[r];
      Plds[prow * 48 + lm + 16] = (__bf16)p1v[r];
    }
    __syncthreads();
    const v16bf pfrag = cat8(*(const v8bf*)&Plds[lm * 48 + hf * 16],
                             *(const v8bf*)&Plds[lm * 48 + hf * 16 + 8]);

    // acc += P(16x32) * V(32x64) ; V column n is row n of Vt (contiguous)
#pragma unroll
    for (int nt = 0; nt < 4; ++nt) {
      const __bf16* vrow = Vh + (size_t)(nt * 16 + lm) * Tc + col0 + hf * 16;
      acc[nt] = wmma_bf16(pfrag, *(const v16bf*)vrow, acc[nt]);
    }
  }

#pragma unroll
  for (int r = 0; r < 8; ++r) {
    const int grow = qrow0 + r + 8 * hf;
    if (grow < Tq) {
      const float inv = 1.f / lrow[r];
#pragma unroll
      for (int nt = 0; nt < 4; ++nt)
        Yh[(size_t)grow * Dc + nt * 16 + lm] = acc[nt][r] * inv;
    }
  }
}

// ---------------------------------------------------------------------------
// Elementwise helpers
// ---------------------------------------------------------------------------
__global__ void transpose_w_kernel(const float* __restrict__ W,
                                   __bf16* __restrict__ WT, int K, int N) {
  const int idx = blockIdx.x * blockDim.x + threadIdx.x;
  if (idx >= N * K) return;
  const int n = idx / K, k = idx % K;
  WT[idx] = (__bf16)W[(size_t)k * N + n];
}

// VT[b][h][dd][t] = x[b][t][h*d+dd]
__global__ void make_vt_kernel(const float* __restrict__ x, __bf16* __restrict__ VT) {
  const int idx = blockIdx.x * blockDim.x + threadIdx.x;
  if (idx >= Bc * Hc * Dc * Tc) return;
  const int t = idx % Tc;
  int r = idx / Tc;
  const int dd = r % Dc; r /= Dc;
  const int h = r % Hc;  const int b = r / Hc;
  VT[idx] = (__bf16)x[((size_t)b * Tc + t) * Cc + h * Dc + dd];
}

// ET[b][h][dd][t] = t<T-1 ? x[b][t+1][h*d+dd] - Y[b][h][t][dd] : 0
__global__ void make_et_kernel(const float* __restrict__ x,
                               const float* __restrict__ Y,
                               __bf16* __restrict__ ET) {
  const int idx = blockIdx.x * blockDim.x + threadIdx.x;
  if (idx >= Bc * Hc * Dc * Tc) return;
  const int t = idx % Tc;
  int r = idx / Tc;
  const int dd = r % Dc; r /= Dc;
  const int h = r % Hc;  const int b = r / Hc;
  float v = 0.f;
  if (t < Tc - 1)
    v = x[((size_t)b * Tc + (t + 1)) * Cc + h * Dc + dd] -
        Y[(((size_t)(b * Hc + h)) * Tc + t) * Dc + dd];
  ET[idx] = (__bf16)v;
}

// yc[b][t][c] = Y[b,h,t,dd] + (t>0 ? Y2[b,h,t-1,dd] : 0)
__global__ void combine_y_kernel(const float* __restrict__ Y,
                                 const float* __restrict__ Y2,
                                 __bf16* __restrict__ yc) {
  const int idx = blockIdx.x * blockDim.x + threadIdx.x;
  if (idx >= Bc * Tc * Cc) return;
  const int c = idx % Cc;
  const int t = (idx / Cc) % Tc;
  const int b = idx / (Cc * Tc);
  const int h = c / Dc, dd = c % Dc;
  float v = Y[(((size_t)(b * Hc + h)) * Tc + t) * Dc + dd];
  if (t > 0) v += Y2[(((size_t)(b * Hc + h)) * Tc + (t - 1)) * Dc + dd];
  yc[idx] = (__bf16)v;
}

// ---------------------------------------------------------------------------
extern "C" void kernel_launch(void* const* d_in, const int* in_sizes, int n_in,
                              void* d_out, int out_size, void* d_ws, size_t ws_size,
                              hipStream_t stream) {
  const float* x      = (const float*)d_in[0];
  const float* W_attn = (const float*)d_in[1];
  const float* b_attn = (const float*)d_in[2];
  const float* W_k2   = (const float*)d_in[3];
  const float* b_k2   = (const float*)d_in[4];
  const float* W_proj = (const float*)d_in[5];
  const float* b_proj = (const float*)d_in[6];
  float* out = (float*)d_out;

  const size_t nHead = (size_t)Bc * Hc * Tc * Dc;   // 4M elems

  size_t off = 0;
  auto wsa = [&](size_t bytes) -> char* {
    char* p = (char*)d_ws + off;
    off = (off + bytes + 255) & ~(size_t)255;
    return p;
  };

  __bf16* WTattn = (__bf16*)wsa((size_t)2 * Cc * Cc * 2);
  __bf16* WTk2   = (__bf16*)wsa((size_t)Cc * Cc * 2);
  __bf16* WTproj = (__bf16*)wsa((size_t)Cc * Cc * 2);
  __bf16* qbf    = (__bf16*)wsa(nHead * 2 + 256);   // +pad: attn2 reads row T
  __bf16* kbf    = (__bf16*)wsa(nHead * 2);
  __bf16* vt     = (__bf16*)wsa(nHead * 2);
  __bf16* k2bf   = (__bf16*)wsa(nHead * 2);         // row T-1 zeroed
  __bf16* et     = (__bf16*)wsa(nHead * 2);         // row T-1 zeroed
  float*  y      = (float*)wsa(nHead * 4);
  float*  y2     = (float*)wsa(nHead * 4);
  __bf16* ycomb  = (__bf16*)wsa((size_t)Bc * Tc * Cc * 2);
  (void)ws_size; (void)in_sizes; (void)n_in; (void)out_size;

  const int mblocks = (Bc * Tc) / 64;               // 64 (64 rows / workgroup)

  // 1) weight transposes -> bf16 (N,K)
  {
    int tot = 2 * Cc * Cc;
    transpose_w_kernel<<<(tot + 255) / 256, 256, 0, stream>>>(W_attn, WTattn, Cc, 2 * Cc);
    tot = Cc * Cc;
    transpose_w_kernel<<<(tot + 255) / 256, 256, 0, stream>>>(W_k2, WTk2, Cc, Cc);
    transpose_w_kernel<<<(tot + 255) / 256, 256, 0, stream>>>(W_proj, WTproj, Cc, Cc);
  }
  // 2) V^T (also v for attn1)
  {
    const int tot = (int)nHead;
    make_vt_kernel<<<(tot + 255) / 256, 256, 0, stream>>>(x, vt);
  }
  // 3) qk = x @ W_attn + b  -> q(scaled), k
  gemm_wmma_kernel<float, 0><<<dim3(mblocks, (2 * Cc) / 64), 128, 0, stream>>>(
      x, WTattn, b_attn, qbf, kbf, Cc, 2 * Cc);
  // 4) k2 = x @ W_k2 + b
  gemm_wmma_kernel<float, 1><<<dim3(mblocks, Cc / 64), 128, 0, stream>>>(
      x, WTk2, b_k2, k2bf, nullptr, Cc, Cc);
  // 5) y = causal_attn(q, k, v)
  flash_attn_kernel<<<dim3(Tc / 16, Bc * Hc), 32, 0, stream>>>(
      qbf, kbf, vt, y, Tc, Tc);
  // 6) e^T = (v[1:] - y[:-1])^T
  {
    const int tot = (int)nHead;
    make_et_kernel<<<(tot + 255) / 256, 256, 0, stream>>>(x, y, et);
  }
  // 7) y2 = causal_attn(q[1:], k2, e)   (length T-1)
  flash_attn_kernel<<<dim3((Tc - 1 + 15) / 16, Bc * Hc), 32, 0, stream>>>(
      qbf + Dc, k2bf, et, y2, Tc - 1, Tc - 1);
  // 8) ycomb = y + shift(y2)  (bf16, (B,T,C))
  {
    const int tot = Bc * Tc * Cc;
    combine_y_kernel<<<(tot + 255) / 256, 256, 0, stream>>>(y, y2, ycomb);
  }
  // 9) out = ycomb @ W_proj + 2*b_proj
  gemm_wmma_kernel<__bf16, 2><<<dim3(mblocks, Cc / 64), 128, 0, stream>>>(
      ycomb, WTproj, b_proj, out, nullptr, Cc, Cc);
}
